// HGCN_18872086299319
// MI455X (gfx1250) — matrix operand order repaired and checked by reading, not compiled, not run on hardware
//
#include <hip/hip_runtime.h>
#include <math.h>

// ---------------------------------------------------------------------------
// HGCN temporal pipeline for MI455X (gfx1250, wave32).
// Dominant op: SUP = adj(8192x8192,f32) @ HT(8192x384) via
// v_wmma_f32_16x16x32_bf16 with hi/lo bf16 error-compensated split.
// adj tiles: TDM (tensor_load_to_lds) double-buffered in LDS, converted
// once per block into fragment-swizzled bf16 LDS tiles.
// ---------------------------------------------------------------------------

typedef __attribute__((ext_vector_type(16))) __bf16        bfx16;
typedef __attribute__((ext_vector_type(8)))  float         fx8;
typedef __attribute__((ext_vector_type(4)))  unsigned int  u32x4;
typedef __attribute__((ext_vector_type(8)))  int           i32x8;
typedef __attribute__((ext_vector_type(4)))  int           i32x4;

#define MINN 1e-15f
#define PMAX 0.996f      // (1 - 4e-3)/sqrt(c), c = 1

__device__ __forceinline__ float wred(float v) {
    #pragma unroll
    for (int m = 16; m > 0; m >>= 1) v += __shfl_xor(v, m, 32);
    return v;
}

__device__ __forceinline__ float artanh_(float x) {
    x = fminf(fmaxf(x, -1.f + 1e-7f), 1.f - 1e-7f);
    return 0.5f * logf((1.f + x) / (1.f - x));
}

// ------------------------- temporal attention ------------------------------
// lhs_pre[t,f] = sum_n U1[n] * x[n,f,t]
__global__ __launch_bounds__(256)
void k_att_lhspre(const float* __restrict__ x, const float* __restrict__ U1,
                  float* __restrict__ lhs_pre, int N, int F, int T) {
    __shared__ float red[256];
    int f = blockIdx.x, t = blockIdx.y, tid = threadIdx.x;
    float s = 0.f;
    for (int n = tid; n < N; n += 256)
        s += U1[n] * x[(size_t)n * F * T + (size_t)f * T + t];
    red[tid] = s; __syncthreads();
    for (int st = 128; st > 0; st >>= 1) {
        if (tid < st) red[tid] += red[tid + st];
        __syncthreads();
    }
    if (tid == 0) lhs_pre[t * F + f] = red[0];
}

// lhs[t,n] = sum_f lhs_pre[t,f] * U2[f,n]
__global__ __launch_bounds__(256)
void k_att_lhs(const float* __restrict__ lhs_pre, const float* __restrict__ U2,
               float* __restrict__ lhs, int N, int F, int T) {
    int n = blockIdx.x * 256 + threadIdx.x;
    int t = blockIdx.y;
    float s = 0.f;
    for (int f = 0; f < F; ++f) s += lhs_pre[t * F + f] * U2[(size_t)f * N + n];
    lhs[(size_t)t * N + n] = s;
}

// rhs[n,t] = sum_f U3[f] * x[n,f,t]
__global__ __launch_bounds__(256)
void k_att_rhs(const float* __restrict__ x, const float* __restrict__ U3,
               float* __restrict__ rhs, int N, int F, int T) {
    int idx = blockIdx.x * 256 + threadIdx.x;
    int n = idx / T, t = idx % T;
    float s = 0.f;
    for (int f = 0; f < F; ++f) s += U3[f] * x[(size_t)n * F * T + (size_t)f * T + t];
    rhs[(size_t)n * T + t] = s;
}

// prod[t,s] = sum_n lhs[t,n] * rhs[n,s]
__global__ __launch_bounds__(256)
void k_att_prod(const float* __restrict__ lhs, const float* __restrict__ rhs,
                float* __restrict__ prod, int N, int T) {
    __shared__ float red[256];
    int t = blockIdx.x / T, sI = blockIdx.x % T, tid = threadIdx.x;
    float s = 0.f;
    for (int n = tid; n < N; n += 256)
        s += lhs[(size_t)t * N + n] * rhs[(size_t)n * T + sI];
    red[tid] = s; __syncthreads();
    for (int st = 128; st > 0; st >>= 1) {
        if (tid < st) red[tid] += red[tid + st];
        __syncthreads();
    }
    if (tid == 0) prod[t * T + sI] = red[0];
}

// E[t,r] = softmax_t( sum_s Ve[t,s] * sigmoid(prod[s,r] + be[s,r]) )
__global__ void k_att_E(const float* __restrict__ prod, const float* __restrict__ be,
                        const float* __restrict__ Ve, float* __restrict__ Eout) {
    __shared__ float P[144], Eb[144], mx[12], sm[12];
    int tid = threadIdx.x;                 // 0..143
    P[tid] = 1.f / (1.f + expf(-(prod[tid] + be[tid])));
    __syncthreads();
    int t = tid / 12, r = tid % 12;
    float s = 0.f;
    for (int k = 0; k < 12; ++k) s += Ve[t * 12 + k] * P[k * 12 + r];
    Eb[tid] = s;
    __syncthreads();
    if (tid < 12) {
        float m = -1e30f;
        for (int k = 0; k < 12; ++k) m = fmaxf(m, Eb[k * 12 + tid]);
        float z = 0.f;
        for (int k = 0; k < 12; ++k) z += expf(Eb[k * 12 + tid] - m);
        mx[tid] = m; sm[tid] = 1.f / z;
    }
    __syncthreads();
    Eout[tid] = expf(Eb[tid] - mx[r]) * sm[r];
}

// --------------- fused (x @ E) + Conv2d(T,T,(1,3),pad (0,1)) ---------------
// in : xin[n, f, t]   out: out[t_o, n, f]
__global__ __launch_bounds__(256)
void k_xform_conv(const float* __restrict__ xin, const float* __restrict__ E,
                  const float* __restrict__ cw, const float* __restrict__ cb,
                  float* __restrict__ out, int N, int F, int T) {
    __shared__ float Es[144];
    __shared__ float Ws[432];
    __shared__ float Bs[12];
    __shared__ float tile[3264];           // rows*(F+2)*12  (<= 8*34*12)
    int tid = threadIdx.x;
    if (tid < 144) Es[tid] = E[tid];
    for (int i = tid; i < 432; i += 256) Ws[i] = cw[i];
    if (tid < 12) Bs[tid] = cb[tid];
    int rows = 256 / F;
    int r = tid / F, f = tid % F;
    int n = blockIdx.x * rows + r;
    int stride = (F + 2) * 12;
    if (f == 0) {
        for (int t = 0; t < 12; ++t) {
            tile[r * stride + t] = 0.f;
            tile[r * stride + (F + 1) * 12 + t] = 0.f;
        }
    }
    float xv[12];
    const float* xp = xin + ((size_t)n * F + f) * 12;
    #pragma unroll
    for (int t = 0; t < 12; ++t) xv[t] = xp[t];
    __syncthreads();                       // Es ready
    #pragma unroll
    for (int tp = 0; tp < 12; ++tp) {
        float s = 0.f;
        #pragma unroll
        for (int t = 0; t < 12; ++t) s += xv[t] * Es[t * 12 + tp];
        tile[r * stride + (f + 1) * 12 + tp] = s;
    }
    __syncthreads();                       // tile ready
    const float* Lm = &tile[r * stride + f * 12];
    const float* Lc = Lm + 12;
    const float* Lp = Lm + 24;
    #pragma unroll
    for (int to = 0; to < 12; ++to) {
        float s = Bs[to];
        #pragma unroll
        for (int ti = 0; ti < 12; ++ti) {
            const float* w = &Ws[(to * 12 + ti) * 3];
            s += Lm[ti] * w[0] + Lc[ti] * w[1] + Lp[ti] * w[2];
        }
        out[((size_t)to * N + n) * F + f] = s;
    }
}

// ---- fused expmap0/proj + mobius_matvec + bias mobius_add + logmap0 -------
// in : xc[t, n, f]  (F=64)   out: HT hi/lo, column-major [col=t*32+d][n], bf16
__global__ __launch_bounds__(256)
void k_hyp_pre(const float* __restrict__ xc, const float* __restrict__ W,
               const float* __restrict__ b, __bf16* __restrict__ Bhi,
               __bf16* __restrict__ Blo, int N, int T) {
    __shared__ float Ws[32 * 65];
    int tid = threadIdx.x;
    for (int i = tid; i < 32 * 64; i += 256)
        Ws[(i >> 6) * 65 + (i & 63)] = W[i];
    __syncthreads();
    int lane = tid & 31, w = tid >> 5;
    int t = blockIdx.y;
    int n = blockIdx.x * 8 + w;
    const float* row = xc + ((size_t)t * N + n) * 64;
    float y0 = row[lane], y1 = row[lane + 32];
    // expmap0 + proj over F=64
    float yn = sqrtf(wred(y0 * y0 + y1 * y1));
    float un = fmaxf(yn, MINN);
    float sc1 = tanhf(un) / un;
    float xh0 = sc1 * y0, xh1 = sc1 * y1;
    float nx = sc1 * yn;
    if (nx > PMAX) { float s = PMAX / nx; xh0 *= s; xh1 *= s; nx = PMAX; }
    float xn = fmaxf(nx, MINN);
    // mobius_matvec: mx[d] = sum_f xh[f] * W[d,f]  (lane = d)
    float mx = 0.f;
    const float* wr = &Ws[lane * 65];
    #pragma unroll
    for (int f = 0; f < 32; ++f) {
        float a0 = __shfl(xh0, f, 32);
        float a1 = __shfl(xh1, f, 32);
        mx += a0 * wr[f] + a1 * wr[f + 32];
    }
    float mxn2 = wred(mx * mx);
    float mxn_raw = sqrtf(mxn2);
    float mxn = fmaxf(mxn_raw, MINN);
    float arg = (mxn / xn) * artanh_(xn);
    float res = tanhf(arg) * mx / mxn;
    float rn = tanhf(arg) * mxn_raw / mxn;
    if (mxn_raw == 0.f) { res = 0.f; rn = 0.f; }
    if (rn > PMAX) { res *= PMAX / rn; }
    // bias_h = proj(expmap0(b))
    float bd = b[lane];
    float bn_ = sqrtf(wred(bd * bd));
    float ub = fmaxf(bn_, MINN);
    float scb = tanhf(ub) / ub;
    float bh = scb * bd;
    float bhn = scb * bn_;
    if (bhn > PMAX) bh *= PMAX / bhn;
    // h = proj(mobius_add(res, bh))
    float x2s = wred(res * res);
    float y2s = wred(bh * bh);
    float xys = wred(res * bh);
    float num = (1.f + 2.f * xys + y2s) * res + (1.f - x2s) * bh;
    float den = fmaxf(1.f + 2.f * xys + x2s * y2s, MINN);
    float hv = num / den;
    float hn = sqrtf(wred(hv * hv));
    if (hn > PMAX) { hv *= PMAX / hn; hn = PMAX; }
    // ht = logmap0(h)
    float pn = fmaxf(hn, MINN);
    float ht = artanh_(pn) * hv / pn;
    // split hi/lo bf16, store column-major for WMMA-B contiguity
    __bf16 hi = (__bf16)ht;
    __bf16 lo = (__bf16)(ht - (float)hi);
    size_t col = (size_t)t * 32 + lane;
    Bhi[col * N + n] = hi;
    Blo[col * N + n] = lo;
}

// ------------- TDM: DMA a 64x32 f32 tile into LDS with stride-33 padding ---
// D# per CDNA5 ISA: count=1, type=2(image),
// data_size=4B, tile 32(x) x 64(y), pad 1 DWORD every 32 DWORDs.
// Toolchain carries the 6-arg builtin: (u32x4, i32x8, i32x4, i32x4, i32x8, i32).
__device__ __forceinline__ void tdm_load_tile(const float* gsrc, unsigned lds_off,
                                              int NN) {
    unsigned long long ga = (unsigned long long)(size_t)gsrc;
    u32x4 g0;
    g0[0] = 1u;                                              // count=1
    g0[1] = lds_off;                                         // lds_addr (bytes)
    g0[2] = (unsigned)(ga & 0xFFFFFFFFu);                    // global_addr[31:0]
    g0[3] = (unsigned)((ga >> 32) & 0x01FFFFFFu) | (2u << 30); // [56:32] | type=2
    i32x8 g1;
    g1[0] = (int)((2u << 16) | (1u << 20) | (4u << 22));     // 4B, pad_en, every 32dw
    g1[1] = (int)((unsigned)(NN & 0xFFFF) << 16);            // tensor_dim0[15:0]
    g1[2] = (int)(((unsigned)NN >> 16) | ((unsigned)(NN & 0xFFFF) << 16)); // dim0 hi | dim1 lo
    g1[3] = (int)(((unsigned)NN >> 16) | (32u << 16));       // dim1 hi | tile_dim0=32
    g1[4] = 64;                                              // tile_dim1=64
    g1[5] = NN;                                              // tensor_dim0_stride lo
    g1[6] = 0;
    g1[7] = 0;
    i32x4 z4 = {0, 0, 0, 0};
    i32x8 z8 = {0, 0, 0, 0, 0, 0, 0, 0};
    __builtin_amdgcn_tensor_load_to_lds(g0, g1, z4, z4, z8, 0);
}

// ---------------------- SUP = adj @ HT  (WMMA bf16 split) ------------------
// block = 8 waves, rows [m0, m0+64) x all 384 cols.
// wave -> 3 col-tiles x 4 m-tiles (B fragments reused across m-tiles).
// adj tiles: TDM double-buffered f32 in LDS; cooperatively split once into
// fragment-swizzled bf16 hi/lo LDS tiles so each lane's A fragment is one
// contiguous 32B LDS read. B: L2-resident global_load_b128.
__global__ __launch_bounds__(256)
void k_adj_gemm(const float* __restrict__ adj, const __bf16* __restrict__ Bhi,
                const __bf16* __restrict__ Blo, float* __restrict__ SUP,
                int NN, int Ncols) {
    __shared__ float  As[2][64 * 33];
    __shared__ __bf16 Ahs[4 * 32 * 16];    // fragment-swizzled hi tile
    __shared__ __bf16 Als[4 * 32 * 16];    // fragment-swizzled lo tile
    int tid = threadIdx.x;
    int lane = tid & 31, wave = tid >> 5;
    int m0 = blockIdx.x * 64;
    fx8 acc[4][3] = {};
    int colt0 = wave * 48;
    int bn    = lane & 15;
    int khalf = (lane >> 4) * 16;
    // conversion mapping: thread -> one 8-element k-run of one row
    // frag element (m,k): lane = (m&15) + 16*(k>>4), elem = k&15
    int cm    = tid >> 2;                  // row 0..63
    int ckb   = (tid & 3) * 8;             // k base 0,8,16,24
    int clane = (cm & 15) + ((ckb >> 4) << 4);
    int cdst  = (((cm >> 4) * 32 + clane) << 4) + (ckb & 15);
    int csrc  = cm * 33 + ckb;
    int cur = 0;
    if (wave == 0) {
        tdm_load_tile(adj + (size_t)m0 * NN, (unsigned)(size_t)&As[0][0], NN);
        __builtin_amdgcn_s_wait_tensorcnt(0);
    }
    __syncthreads();
    for (int k0 = 0; k0 < NN; k0 += 32) {
        // cooperative f32 -> bf16 hi/lo split (once per block, not per wave)
        #pragma unroll
        for (int j = 0; j < 8; ++j) {
            float a = As[cur][csrc + j];
            __bf16 h = (__bf16)a;
            Ahs[cdst + j] = h;
            Als[cdst + j] = (__bf16)(a - (float)h);
        }
        __syncthreads();                   // bf16 fragment tiles ready
        if (wave == 0 && k0 + 32 < NN)     // TDM prefetch next f32 tile
            tdm_load_tile(adj + (size_t)m0 * NN + k0 + 32,
                          (unsigned)(size_t)&As[cur ^ 1][0], NN);
        // B fragments for this K-step (k = 16*(lane>>4) + elem, same order as A), reused 4x
        bfx16 bhf[3], blf[3];
        #pragma unroll
        for (int tix = 0; tix < 3; ++tix) {
            size_t coff = (size_t)(colt0 + tix * 16 + bn) * NN + k0 + khalf;
            bhf[tix] = *(const bfx16*)(Bhi + coff);
            blf[tix] = *(const bfx16*)(Blo + coff);
        }
        #pragma unroll
        for (int mt = 0; mt < 4; ++mt) {
            bfx16 Ahi = *(const bfx16*)(&Ahs[(mt * 32 + lane) << 4]);
            bfx16 Alo = *(const bfx16*)(&Als[(mt * 32 + lane) << 4]);
            #pragma unroll
            for (int tix = 0; tix < 3; ++tix) {
                acc[mt][tix] = __builtin_amdgcn_wmma_f32_16x16x32_bf16(
                    false, Ahi, false, bhf[tix], (short)0, acc[mt][tix], false, false);
                acc[mt][tix] = __builtin_amdgcn_wmma_f32_16x16x32_bf16(
                    false, Ahi, false, blf[tix], (short)0, acc[mt][tix], false, false);
                acc[mt][tix] = __builtin_amdgcn_wmma_f32_16x16x32_bf16(
                    false, Alo, false, bhf[tix], (short)0, acc[mt][tix], false, false);
            }
        }
        if (wave == 0) __builtin_amdgcn_s_wait_tensorcnt(0);
        __syncthreads();                   // next tile landed; LDS reads done
        cur ^= 1;
    }
    // epilogue: 16x16 f32 C layout: VGPR r, lanes0-15 M=r, lanes16-31 M=r+8
    int rb = (lane >> 4) << 3;
    int cn = lane & 15;
    #pragma unroll
    for (int mt = 0; mt < 4; ++mt) {
        #pragma unroll
        for (int tix = 0; tix < 3; ++tix) {
            int col = colt0 + tix * 16 + cn;
            #pragma unroll
            for (int r = 0; r < 8; ++r)
                SUP[(size_t)(m0 + mt * 16 + rb + r) * Ncols + col] = acc[mt][tix][r];
        }
    }
}

// ------------- proj(expmap0(sup)) -> relu(logmap0) -> proj(expmap0) --------
// in : SUP[n, t*32+d]   out: h[n, d, t]
__global__ __launch_bounds__(256)
void k_hyp_post(const float* __restrict__ SUP, float* __restrict__ h,
                int N, int T) {
    int tid = threadIdx.x;
    int lane = tid & 31, w = tid >> 5;
    int t = blockIdx.y;
    int n = blockIdx.x * 8 + w;
    float s = SUP[(size_t)n * 384 + t * 32 + lane];
    float sn = sqrtf(wred(s * s));
    float un = fmaxf(sn, MINN);
    float sc = tanhf(un) / un;
    float v = sc * s;
    float nv = sc * sn;
    if (nv > PMAX) { v *= PMAX / nv; nv = PMAX; }
    float pn = fmaxf(nv, MINN);
    float u = artanh_(pn) * v / pn;
    u = fmaxf(u, 0.f);                               // relu in tangent space
    float unr = sqrtf(wred(u * u));
    float un2 = fmaxf(unr, MINN);
    float sc2 = tanhf(un2) / un2;
    float o = sc2 * u;
    float no = sc2 * unr;
    if (no > PMAX) o *= PMAX / no;
    h[((size_t)n * 32 + lane) * 12 + t] = o;
}

// ---------------------------------------------------------------------------
extern "C" void kernel_launch(void* const* d_in, const int* in_sizes, int n_in,
                              void* d_out, int out_size, void* d_ws, size_t ws_size,
                              hipStream_t stream) {
    const float* x   = (const float*)d_in[0];
    const float* adj = (const float*)d_in[1];
    const float* U1a = (const float*)d_in[2];
    const float* U2a = (const float*)d_in[3];
    const float* U3a = (const float*)d_in[4];
    const float* bea = (const float*)d_in[5];
    const float* Vea = (const float*)d_in[6];
    const float* U1b = (const float*)d_in[7];
    const float* U2b = (const float*)d_in[8];
    const float* U3b = (const float*)d_in[9];
    const float* beb = (const float*)d_in[10];
    const float* Veb = (const float*)d_in[11];
    const float* W   = (const float*)d_in[12];
    const float* b   = (const float*)d_in[13];
    const float* c1w = (const float*)d_in[14];
    const float* c1b = (const float*)d_in[15];
    const float* c2w = (const float*)d_in[16];
    const float* c2b = (const float*)d_in[17];
    const int N = 8192, T = 12, F = 64, D = 32;

    char* ws = (char*)d_ws;
    float*  xc   = (float*)ws;                       // [T,N,F]   25.2 MB
    float*  SUP  = xc;                               // overlay (xc dead by GEMM)
    float*  hbuf = xc + 3145728;                     // [N,D,T]   overlay upper half
    __bf16* Bhi  = (__bf16*)(ws + 25165824);         // [384, N]  col-major
    __bf16* Blo  = Bhi + 3145728;
    float*  sm   = (float*)(ws + 25165824 + 2 * 6291456);
    float*  lhs_pre = sm;                            // 768
    float*  lhs     = lhs_pre + 768;                 // T*N
    float*  rhs     = lhs + 98304;                   // N*T
    float*  prod    = rhs + 98304;                   // 144
    float*  E1      = prod + 144;
    float*  E2      = E1 + 144;

    // ---- temporal attention 1 + time conv 1 ----
    k_att_lhspre<<<dim3(F, T), 256, 0, stream>>>(x, U1a, lhs_pre, N, F, T);
    k_att_lhs   <<<dim3(N / 256, T), 256, 0, stream>>>(lhs_pre, U2a, lhs, N, F, T);
    k_att_rhs   <<<dim3(N * T / 256), 256, 0, stream>>>(x, U3a, rhs, N, F, T);
    k_att_prod  <<<dim3(T * T), 256, 0, stream>>>(lhs, rhs, prod, N, T);
    k_att_E     <<<1, T * T, 0, stream>>>(prod, bea, Vea, E1);
    k_xform_conv<<<dim3(N / (256 / F)), 256, 0, stream>>>(x, E1, c1w, c1b, xc, N, F, T);

    // ---- hyperbolic GCN: per-row mobius math, big WMMA GEMM, per-row tail ----
    k_hyp_pre   <<<dim3(N / 8, T), 256, 0, stream>>>(xc, W, b, Bhi, Blo, N, T);
    k_adj_gemm  <<<dim3(N / 64), 256, 0, stream>>>(adj, Bhi, Blo, SUP, N, T * D);
    k_hyp_post  <<<dim3(N / 8, T), 256, 0, stream>>>(SUP, hbuf, N, T);

    // ---- temporal attention 2 + time conv 2 ----
    k_att_lhspre<<<dim3(D, T), 256, 0, stream>>>(hbuf, U1b, lhs_pre, N, D, T);
    k_att_lhs   <<<dim3(N / 256, T), 256, 0, stream>>>(lhs_pre, U2b, lhs, N, D, T);
    k_att_rhs   <<<dim3(N * T / 256), 256, 0, stream>>>(hbuf, U3b, rhs, N, D, T);
    k_att_prod  <<<dim3(T * T), 256, 0, stream>>>(lhs, rhs, prod, N, T);
    k_att_E     <<<1, T * T, 0, stream>>>(prod, beb, Veb, E2);
    k_xform_conv<<<dim3(N / (256 / D)), 256, 0, stream>>>(hbuf, E2, c2w, c2b,
                                                          (float*)d_out, N, D, T);
}
